// RFNOTF_80066780332639
// MI455X (gfx1250) — compile-verified, matrix operand order checked by reading, not run on hardware
//
#include <hip/hip_runtime.h>

// FNO spectral conv via 4 skinny DFT GEMMs on v_wmma_f32_16x16x32_f16 (wave32).
// B=2, H=W=1024, C=32, modes 64x32 (k1 in {0..31, 992..1023}, k2 in {0..31}).

typedef __attribute__((ext_vector_type(16))) _Float16 v16h;
typedef __attribute__((ext_vector_type(8)))  float    v8f;
typedef _Float16 half_t;

#define DEV __device__ __forceinline__

static constexpr int Bn = 2, Hn = 1024, Wn = 1024, Cn = 32;
static constexpr int KSPLIT = 4;  // GEMM2 K-parallelism (1024/4 = 256 per slice)

// Workspace layout (bytes). Total ~26 MB.
static constexpr size_t OFF_F1T = 0;                         // [64 j][1024 w] f16
static constexpr size_t OFF_F2  = OFF_F1T + 64 * 1024 * 2;   // [128 r][1024 h] f16
static constexpr size_t OFF_F3  = OFF_F2  + 128 * 1024 * 2;  // [1024 h][128 j] f16
static constexpr size_t OFF_F4T = OFF_F3  + 1024 * 128 * 2;  // [1024 w][64 j] f16
static constexpr size_t OFF_TT  = OFF_F4T + 1024 * 64 * 2;   // [B][C][64 j][1024 h] f16
static constexpr size_t OFF_P   = OFF_TT  + (size_t)Bn * Cn * 64 * 1024 * 2;            // [KSPLIT][B][C][128][64] f32
static constexpr size_t OFF_ZT  = OFF_P   + (size_t)KSPLIT * Bn * Cn * 128 * 64 * 4;    // [B][C][64 n][128 k] f16
static constexpr size_t OFF_U   = OFF_ZT  + (size_t)Bn * Cn * 64 * 128 * 2;             // [B][H][C][64 j] f16

DEV v8f wmma(v16h a, v16h b, v8f c) {
  // D = A(16x32 f16) x B(32x16 f16) + C(16x16 f32)
  return __builtin_amdgcn_wmma_f32_16x16x32_f16(
      /*neg_a=*/false, a, /*neg_b=*/false, b,
      /*c_mod=*/(short)0, c, /*reuse_a=*/false, /*reuse_b=*/false);
}

// ---------------------------------------------------------------------------
// Twiddle tables. Phase (k*n) mod 1024 computed exactly in integers.
// ---------------------------------------------------------------------------
__global__ void k_init(half_t* __restrict__ f1t, half_t* __restrict__ f2,
                       half_t* __restrict__ f3, half_t* __restrict__ f4t) {
  const int t = blockIdx.x * 256 + threadIdx.x;
  const float TP = 6.28318530717958647692f / 1024.0f;
  if (t < 64 * 1024) {                 // F1t[j][w]: j<32 cos, j>=32 -sin (e^{-i})
    int j = t >> 10, w = t & 1023;
    int ph = ((j & 31) * w) & 1023;
    float ang = TP * (float)ph;
    f1t[t] = (half_t)((j < 32) ? __cosf(ang) : -__sinf(ang));
  }
  if (t < 128 * 1024) {                // F2[r][h]: rows 0..63 cos, 64..127 +sin
    int r = t >> 10, h = t & 1023;
    int rr = r & 63;
    int k1 = (rr < 32) ? rr : 992 + (rr - 32);
    int ph = (k1 * h) & 1023;
    float ang = TP * (float)ph;
    f2[t] = (half_t)((r < 64) ? __cosf(ang) : __sinf(ang));
  }
  if (t < 1024 * 128) {                // F3[h][j]: j<64 cos, j>=64 +sin; /1024
    int h = t >> 7, j = t & 127;
    int jj = j & 63;
    int k1 = (jj < 32) ? jj : 992 + (jj - 32);
    int ph = (k1 * h) & 1023;
    float ang = TP * (float)ph;
    float v = ((j < 64) ? __cosf(ang) : __sinf(ang)) * (1.0f / 1024.0f);
    f3[t] = (half_t)v;
  }
  if (t < 1024 * 64) {                 // F4t[w][j]: j<32 a*cos, j>=32 -a*sin; /1024
    int w = t >> 6, j = t & 63;
    int k2 = j & 31;
    int ph = (k2 * w) & 1023;
    float ang = TP * (float)ph;
    float alpha = (k2 == 0) ? 1.0f : 2.0f;  // Hermitian doubling
    float v = ((j < 32) ? __cosf(ang) : -__sinf(ang)) * (alpha / 1024.0f);
    f4t[t] = (half_t)v;
  }
}

// ---------------------------------------------------------------------------
// GEMM1: per (b,h) slice: T[32c x 64j] = x_sliceT[32c x 1024w] @ F1[1024w x 64j]
// Software-pipelined: next 8KB x-chunk (K=64) loads in flight during the two
// WMMAs of the current chunk. Output scattered into Tt[b][c][j][h].
// ---------------------------------------------------------------------------
__global__ void k_gemm1(const float* __restrict__ x, const half_t* __restrict__ f1t,
                        half_t* __restrict__ tt) {
  const int bh = blockIdx.x;
  const int b = bh >> 10, h = bh & 1023;
  const float* xs = x + (size_t)bh * (Wn * Cn);

  __shared__ __align__(32) half_t Atile[32 * 64];  // [c][w_local 0..63]

  const int lane = threadIdx.x & 31;
  const int wave = threadIdx.x >> 5;     // 8 waves
  const int mt = wave & 1;               // c-tile (0..1)
  const int nt = wave >> 1;              // j-tile (0..3)
  const int frow = lane & 15, fk = (lane >> 4) * 16;

  const int wl0 = threadIdx.x >> 3;            // w_local of first float4
  const int c0 = (threadIdx.x & 7) * 4;        // channel base of this thread

  float4 r0, r1;  // prefetch registers: 64w x 32c chunk = 2 float4 per thread
  {
    const float4* src = (const float4*)xs;
    r0 = src[threadIdx.x];
    r1 = src[threadIdx.x + 256];
  }

  v8f acc = {};
  for (int kb = 0; kb < 1024; kb += 64) {
    __syncthreads();  // previous chunk's frag reads complete
    // registers -> LDS (f32 -> f16, transpose to [c][w])
    Atile[(c0 + 0) * 64 + wl0]      = (half_t)r0.x;
    Atile[(c0 + 1) * 64 + wl0]      = (half_t)r0.y;
    Atile[(c0 + 2) * 64 + wl0]      = (half_t)r0.z;
    Atile[(c0 + 3) * 64 + wl0]      = (half_t)r0.w;
    Atile[(c0 + 0) * 64 + wl0 + 32] = (half_t)r1.x;
    Atile[(c0 + 1) * 64 + wl0 + 32] = (half_t)r1.y;
    Atile[(c0 + 2) * 64 + wl0 + 32] = (half_t)r1.z;
    Atile[(c0 + 3) * 64 + wl0 + 32] = (half_t)r1.w;
    __syncthreads();
    if (kb + 64 < 1024) {  // issue next chunk loads; overlap with WMMA below
      const float4* src = (const float4*)(xs + (kb + 64) * 32);
      r0 = src[threadIdx.x];
      r1 = src[threadIdx.x + 256];
    }
    v16h a0 = *(const v16h*)(Atile + (mt * 16 + frow) * 64 + fk);
    v16h a1 = *(const v16h*)(Atile + (mt * 16 + frow) * 64 + 32 + fk);
    v16h b0 = *(const v16h*)(f1t + (size_t)(nt * 16 + frow) * 1024 + kb + fk);
    v16h b1 = *(const v16h*)(f1t + (size_t)(nt * 16 + frow) * 1024 + kb + 32 + fk);
    acc = wmma(a0, b0, acc);
    acc = wmma(a1, b1, acc);
  }
  // store C tile -> Tt[b][c][j][h]
  const int hi8 = (lane >> 4) * 8;
#pragma unroll
  for (int j = 0; j < 8; ++j) {
    int c = mt * 16 + j + hi8;
    int n = nt * 16 + frow;
    tt[(((size_t)(b * Cn + c)) * 64 + n) * 1024 + h] = (half_t)acc[j];
  }
}

// ---------------------------------------------------------------------------
// GEMM2: per (b,c) x K-slice: Pp[s][img][128 x 64] = F2_slice @ T_c_slice
// K split 4 ways (256 each) for occupancy; weights kernel reduces the slices.
// ---------------------------------------------------------------------------
__global__ void k_gemm2(const half_t* __restrict__ f2, const half_t* __restrict__ tt,
                        float* __restrict__ p) {
  const int img = blockIdx.x & 63;        // b*32+c
  const int slice = blockIdx.x >> 6;      // 0..3
  const int kbase = slice * (1024 / KSPLIT);
  const half_t* tb = tt + (size_t)img * 64 * 1024;
  const int lane = threadIdx.x & 31;
  const int wave = threadIdx.x >> 5;      // m-tile 0..7
  const int frow = lane & 15, fk = (lane >> 4) * 16;

  v8f acc[4] = {};
  for (int k = 0; k < 1024 / KSPLIT; k += 32) {
    const int kb = kbase + k;
    v16h a = *(const v16h*)(f2 + (size_t)(wave * 16 + frow) * 1024 + kb + fk);
#pragma unroll
    for (int nt = 0; nt < 4; ++nt) {
      v16h bf = *(const v16h*)(tb + (size_t)(nt * 16 + frow) * 1024 + kb + fk);
      acc[nt] = wmma(a, bf, acc[nt]);
    }
  }
  float* pp = p + ((size_t)slice * 64 + img) * 128 * 64;
  const int hi8 = (lane >> 4) * 8;
#pragma unroll
  for (int nt = 0; nt < 4; ++nt)
#pragma unroll
    for (int j = 0; j < 8; ++j)
      pp[(size_t)(wave * 16 + j + hi8) * 64 + nt * 16 + frow] = acc[nt][j];
}

// ---------------------------------------------------------------------------
// Weights: reduce K-slices of P, form Y, complex mult by w0*(w1|w2), build
// Zstack^T[b][c][64n][128k]: rows k<64: [Zr|Zi]; rows k>=64: [-Zi|Zr].
// ---------------------------------------------------------------------------
__global__ void k_weights(const float* __restrict__ p,
                          const float* __restrict__ w0re, const float* __restrict__ w0im,
                          const float* __restrict__ w1re, const float* __restrict__ w1im,
                          const float* __restrict__ w2re, const float* __restrict__ w2im,
                          half_t* __restrict__ zt) {
  const int img = blockIdx.x;
  const int c = img & 31;
  half_t* z = zt + (size_t)img * 64 * 128;
  const float ar = w0re[c], ai = w0im[c];
  for (int t = threadIdx.x; t < 64 * 32; t += 256) {
    int k1r = t >> 5, k2 = t & 31;
    float yr = 0.f, yi = 0.f;
#pragma unroll
    for (int s = 0; s < KSPLIT; ++s) {
      const float* pp = p + ((size_t)s * 64 + img) * 128 * 64;
      yr += pp[k1r * 64 + k2]      + pp[(64 + k1r) * 64 + 32 + k2];
      yi += pp[k1r * 64 + 32 + k2] - pp[(64 + k1r) * 64 + k2];
    }
    float wr, wi;
    if (k1r < 32) { wr = w1re[(c * 32 + k1r) * 32 + k2];        wi = w1im[(c * 32 + k1r) * 32 + k2]; }
    else          { wr = w2re[(c * 32 + (k1r - 32)) * 32 + k2]; wi = w2im[(c * 32 + (k1r - 32)) * 32 + k2]; }
    float cr = ar * wr - ai * wi, ci = ar * wi + ai * wr;   // w0 * w1/2
    float zr = yr * cr - yi * ci, zi = yr * ci + yi * cr;
    z[(size_t)k2 * 128 + k1r]             = (half_t)zr;
    z[(size_t)(32 + k2) * 128 + k1r]      = (half_t)zi;
    z[(size_t)k2 * 128 + 64 + k1r]        = (half_t)(-zi);
    z[(size_t)(32 + k2) * 128 + 64 + k1r] = (half_t)zr;
  }
}

// ---------------------------------------------------------------------------
// GEMM3: per (b,c,hblock): U[128h x 64] = F3[128h x 128k] @ Zstack[128k x 64]
// Output U[b][h][c][64] (A layout for GEMM4).
// ---------------------------------------------------------------------------
__global__ void k_gemm3(const half_t* __restrict__ f3, const half_t* __restrict__ zt,
                        half_t* __restrict__ u) {
  const int img = blockIdx.x >> 3;
  const int hb = blockIdx.x & 7;
  const int b = img >> 5, c = img & 31;
  const half_t* z = zt + (size_t)img * 64 * 128;
  const int lane = threadIdx.x & 31;
  const int wave = threadIdx.x >> 5;
  const int frow = lane & 15, fk = (lane >> 4) * 16;
  const int hbase = hb * 128;

  v8f acc[4] = {};
#pragma unroll
  for (int kb = 0; kb < 128; kb += 32) {
    v16h a = *(const v16h*)(f3 + (size_t)(hbase + wave * 16 + frow) * 128 + kb + fk);
#pragma unroll
    for (int nt = 0; nt < 4; ++nt) {
      v16h bf = *(const v16h*)(z + (size_t)(nt * 16 + frow) * 128 + kb + fk);
      acc[nt] = wmma(a, bf, acc[nt]);
    }
  }
  const int hi8 = (lane >> 4) * 8;
#pragma unroll
  for (int nt = 0; nt < 4; ++nt)
#pragma unroll
    for (int j = 0; j < 8; ++j) {
      int h = hbase + wave * 16 + j + hi8;
      u[(((size_t)b * 1024 + h) * Cn + c) * 64 + nt * 16 + frow] = (half_t)acc[nt][j];
    }
}

// ---------------------------------------------------------------------------
// GEMM4: per (b,h): y[32c x 1024w] = U_slice[32c x 64j] @ F4[64j x 1024w]
// then transpose via LDS and write NHWC coalesced (128B per wave).
// ---------------------------------------------------------------------------
__global__ void k_gemm4(const half_t* __restrict__ u, const half_t* __restrict__ f4t,
                        float* __restrict__ out) {
  const int bh = blockIdx.x;
  const half_t* us = u + (size_t)bh * Cn * 64;
  float* outp = out + (size_t)bh * (Wn * Cn);
  const int lane = threadIdx.x & 31;
  const int wave = threadIdx.x >> 5;
  const int mt = wave & 1;      // c-tile
  const int ntq = wave >> 1;    // w-tile within 64-wide chunk
  const int frow = lane & 15, fk = (lane >> 4) * 16;
  const int hi8 = (lane >> 4) * 8;

  __shared__ float ylds[64][33];  // [w_local][c], padded

  // A fragments (K=64 -> 2 k-steps) are loop-invariant
  v16h a0 = *(const v16h*)(us + (size_t)(mt * 16 + frow) * 64 + fk);
  v16h a1 = *(const v16h*)(us + (size_t)(mt * 16 + frow) * 64 + 32 + fk);

  for (int wc = 0; wc < 1024; wc += 64) {
    const int nglob = wc + ntq * 16 + frow;
    v16h b0 = *(const v16h*)(f4t + (size_t)nglob * 64 + fk);
    v16h b1 = *(const v16h*)(f4t + (size_t)nglob * 64 + 32 + fk);
    v8f acc = {};
    acc = wmma(a0, b0, acc);
    acc = wmma(a1, b1, acc);
    __syncthreads();  // previous chunk's LDS fully consumed
#pragma unroll
    for (int j = 0; j < 8; ++j)
      ylds[ntq * 16 + frow][mt * 16 + j + hi8] = acc[j];
    __syncthreads();
#pragma unroll
    for (int i = 0; i < 8; ++i) {
      int wl = wave + i * 8;     // 0..63
      outp[(size_t)(wc + wl) * Cn + lane] = ylds[wl][lane];
    }
  }
}

extern "C" void kernel_launch(void* const* d_in, const int* in_sizes, int n_in,
                              void* d_out, int out_size, void* d_ws, size_t ws_size,
                              hipStream_t stream) {
  (void)in_sizes; (void)n_in; (void)out_size; (void)ws_size;
  const float* x    = (const float*)d_in[0];
  const float* w0re = (const float*)d_in[1];
  const float* w0im = (const float*)d_in[2];
  const float* w1re = (const float*)d_in[3];
  const float* w1im = (const float*)d_in[4];
  const float* w2re = (const float*)d_in[5];
  const float* w2im = (const float*)d_in[6];

  char* ws = (char*)d_ws;
  half_t* f1t = (half_t*)(ws + OFF_F1T);
  half_t* f2  = (half_t*)(ws + OFF_F2);
  half_t* f3  = (half_t*)(ws + OFF_F3);
  half_t* f4t = (half_t*)(ws + OFF_F4T);
  half_t* tt  = (half_t*)(ws + OFF_TT);
  float*  p   = (float*) (ws + OFF_P);
  half_t* zt  = (half_t*)(ws + OFF_ZT);
  half_t* uu  = (half_t*)(ws + OFF_U);

  k_init   <<<512, 256, 0, stream>>>(f1t, f2, f3, f4t);
  k_gemm1  <<<Bn * Hn, 256, 0, stream>>>(x, f1t, tt);
  k_gemm2  <<<Bn * Cn * KSPLIT, 256, 0, stream>>>(f2, tt, p);
  k_weights<<<Bn * Cn, 256, 0, stream>>>(p, w0re, w0im, w1re, w1im, w2re, w2im, zt);
  k_gemm3  <<<Bn * Cn * 8, 256, 0, stream>>>(f3, zt, uu);
  k_gemm4  <<<Bn * Hn, 256, 0, stream>>>(uu, f4t, (float*)d_out);
}
